// CasualSelfAttention_53317724012590
// MI455X (gfx1250) — matrix operand order repaired and checked by reading, not compiled, not run on hardware
//
#include <hip/hip_runtime.h>

// ---- problem constants (fixed by the reference) ----
#define BATCH 4
#define SLEN  2048
#define EMB   2048
#define NHEAD 16
#define DHEAD 128

typedef __attribute__((ext_vector_type(16))) __bf16 v16bf;
typedef __attribute__((ext_vector_type(8)))  float  v8f;

union FragAB { v16bf v; uint4 q[2]; unsigned u[8]; };

__device__ __forceinline__ unsigned bf16_1(float f) {
    unsigned u = __float_as_uint(f);
    u += 0x7FFFu + ((u >> 16) & 1u);     // round-to-nearest-even
    return u >> 16;
}
__device__ __forceinline__ unsigned bf16_pk(float lo, float hi) {
    return bf16_1(lo) | (bf16_1(hi) << 16);
}

// ---- CDNA5 async global->LDS copies (ASYNCcnt), with safe fallback ----
#if defined(__gfx1250__) && __has_builtin(__builtin_amdgcn_global_load_async_to_lds_b128)
#define USE_ASYNC 1
#else
#define USE_ASYNC 0
#endif

// The builtin's pointer params are 'int __attribute__((vector_size(16))) *'
typedef int v4i_a __attribute__((vector_size(16)));

__device__ __forceinline__ void cp16_lds(void* lds, const void* g) {
#if USE_ASYNC
    __builtin_amdgcn_global_load_async_to_lds_b128((v4i_a*)g, (v4i_a*)lds, 0, 0);
#else
    *(uint4*)lds = *(const uint4*)g;
#endif
}

#if USE_ASYNC
#  if __has_builtin(__builtin_amdgcn_s_wait_asynccnt)
#    define WAIT_ASYNC(n) __builtin_amdgcn_s_wait_asynccnt(n)
#  else
#    define WAIT_ASYNC(n) asm volatile("s_wait_asynccnt %0" :: "n"(n) : "memory")
#  endif
#else
#  define WAIT_ASYNC(n) ((void)0)
#endif

// ============================================================================
// Pre-pack kernels: fp32 -> bf16 once, outside the hot loops.
// ============================================================================
__global__ __launch_bounds__(256)
void pack_adj_bf16(const float* __restrict__ src, unsigned* __restrict__ dst)
{
    size_t i = (size_t)blockIdx.x * 256 + threadIdx.x;
    dst[i] = bf16_pk(src[2 * i], src[2 * i + 1]);
}

// W[K][N] fp32 -> Wp[K/2][N] dwords, dword = pack(W[2k2][n], W[2k2+1][n])
__global__ __launch_bounds__(256)
void pack_pairs_bf16(const float* __restrict__ src, unsigned* __restrict__ dst, int N)
{
    int n  = blockIdx.x * 256 + threadIdx.x;
    int k2 = blockIdx.y;
    const float* g = src + (size_t)(2 * k2) * N + n;
    dst[(size_t)k2 * N + n] = bf16_pk(g[0], g[N]);
}

// ============================================================================
// GEMM: C[M,N] = A_bf16[M,K] * B_packed[K,N] + bias
// 128x128 tile, 8 waves, K-step 32, double-buffered async LDS loads.
// mode 0 (QKV): scatter k,v fp32 to d_out + q,k bf16 + pair-packed V^T
// mode 1: plain fp32 row-major store
// ============================================================================
__global__ __launch_bounds__(256)
void gemm_bf16_wmma(const unsigned short* __restrict__ A,   // [M][K] bf16
                    const unsigned* __restrict__ Bp,        // [K/2][N] packed
                    const float* __restrict__ bias,
                    float* __restrict__ outF,
                    float* __restrict__ kF, float* __restrict__ vF,
                    unsigned short* __restrict__ qB, unsigned short* __restrict__ kB,
                    unsigned* __restrict__ vtP,
                    int N, int K, int mode)
{
    __shared__ unsigned lA[2][128 * 20];   // 80B-padded rows (bank-conflict free)
    __shared__ unsigned lB[2][16 * 128];   // [k2][n], xor-swizzled on n bit5

    const int tid  = threadIdx.x;
    const int wave = tid >> 5, lane = tid & 31;
    const int wr = wave & 1, wc = wave >> 1;
    const int lane16 = lane & 15, khalf = lane >> 4;
    const int rowbase = blockIdx.y * 128;
    const int colbase = blockIdx.x * 128;

    v8f acc[4][2] = {};

    auto load_tiles = [&](int kc, int buf) {
        #pragma unroll
        for (int i = 0; i < 2; ++i) {                 // A tile 128x32 bf16 (8 KB)
            int p = tid + i * 256;
            int m = p >> 2, c = p & 3;
            cp16_lds((char*)lA[buf] + m * 80 + c * 16,
                     A + (size_t)(rowbase + m) * K + kc + c * 8);
        }
        #pragma unroll
        for (int i = 0; i < 2; ++i) {                 // B tile 16x128 dwords (8 KB)
            int p = tid + i * 256;
            int r = p >> 5, c = p & 31;
            int cs = c ^ ((r >> 3) << 3);
            cp16_lds((char*)lB[buf] + r * 512 + cs * 16,
                     Bp + (size_t)((kc >> 1) + r) * N + colbase + c * 4);
        }
    };

    const int nk = K / 32;
    load_tiles(0, 0);
    for (int t = 0; t < nk; ++t) {
        const int cur = t & 1;
        if (t + 1 < nk) { load_tiles((t + 1) * 32, cur ^ 1); WAIT_ASYNC(4); }
        else            { WAIT_ASYNC(0); }
        __syncthreads();

        FragAB fa[4];
        #pragma unroll
        for (int rt = 0; rt < 4; ++rt) {
            int row = wr * 64 + rt * 16 + lane16;
            const uint4* pa = (const uint4*)&lA[cur][row * 20 + khalf * 8];
            fa[rt].q[0] = pa[0];            // K = khalf*16 .. +7
            fa[rt].q[1] = pa[1];            // K = khalf*16+8 .. +15
        }
        FragAB fb[2];
        #pragma unroll
        for (int ct = 0; ct < 2; ++ct) {
            int n = (wc * 32 + ct * 16 + lane16) ^ (khalf << 5);
            #pragma unroll
            for (int j = 0; j < 8; ++j)
                fb[ct].u[j] = lB[cur][(khalf * 8 + j) * 128 + n];
        }
        #pragma unroll
        for (int rt = 0; rt < 4; ++rt)
            #pragma unroll
            for (int ct = 0; ct < 2; ++ct)
                acc[rt][ct] = __builtin_amdgcn_wmma_f32_16x16x32_bf16(
                    false, fa[rt].v, false, fb[ct].v,
                    (short)0, acc[rt][ct], false, false);
        __syncthreads();
    }

    // ---- epilogue ----
    #pragma unroll
    for (int rt = 0; rt < 4; ++rt) {
        #pragma unroll
        for (int ct = 0; ct < 2; ++ct) {
            int colg = colbase + wc * 32 + ct * 16 + lane16;
            float bv = bias[colg];
            float val[8];
            #pragma unroll
            for (int v = 0; v < 8; ++v) val[v] = acc[rt][ct][v] + bv;
            int rowg = rowbase + wr * 64 + rt * 16 + khalf * 8;   // multiple of 8
            if (mode == 1) {
                #pragma unroll
                for (int v = 0; v < 8; ++v)
                    outF[(size_t)(rowg + v) * N + colg] = val[v];
            } else {
                int sec = colg >> 11;                      // 0=q 1=k 2=v
                int e = colg & (EMB - 1), h = e >> 7, d = e & (DHEAD - 1);
                int b = rowg >> 11, s = rowg & (SLEN - 1);
                size_t base = ((size_t)(b * NHEAD + h) * SLEN + s) * DHEAD + d;
                if (sec == 0) {
                    #pragma unroll
                    for (int v = 0; v < 8; ++v)
                        qB[base + (size_t)v * DHEAD] = (unsigned short)bf16_1(val[v]);
                } else if (sec == 1) {
                    #pragma unroll
                    for (int v = 0; v < 8; ++v) {
                        kF[base + (size_t)v * DHEAD] = val[v];
                        kB[base + (size_t)v * DHEAD] = (unsigned short)bf16_1(val[v]);
                    }
                } else {
                    #pragma unroll
                    for (int v = 0; v < 8; ++v)
                        vF[base + (size_t)v * DHEAD] = val[v];
                    size_t tb = ((size_t)(b * NHEAD + h) * (SLEN / 2) + (s >> 1)) * DHEAD + d;
                    #pragma unroll
                    for (int v2 = 0; v2 < 4; ++v2)          // pair-packed V^T
                        vtP[tb + (size_t)v2 * DHEAD] = bf16_pk(val[2 * v2], val[2 * v2 + 1]);
                }
            }
        }
    }
}

// ============================================================================
// Flash attention, bf16 tiles, double-buffered async K/V streaming.
// ============================================================================
__global__ __launch_bounds__(256)
void flash_attn_wmma(const unsigned short* __restrict__ qB,
                     const unsigned short* __restrict__ kB,
                     const unsigned* __restrict__ vtP,
                     unsigned short* __restrict__ yB)
{
    __shared__ unsigned lK[2][32 * 64];       // [tok][feat-pair], xor-swizzled
    __shared__ unsigned lV[2][16 * 128];      // [tok-pair][feat], xor-swizzled
    __shared__ unsigned short lP[8][16 * 40]; // per-wave P tile, 80B rows

    const int tid  = threadIdx.x;
    const int wave = tid >> 5, lane = tid & 31;
    const int lane16 = lane & 15, khalf = lane >> 4;
    const int qb = blockIdx.x, h = blockIdx.y, b = blockIdx.z;
    const size_t bh = (size_t)(b * NHEAD + h);

    const int q0 = qb * 128;
    const int qrow = q0 + wave * 16 + lane16;
    const float sscale = 0.08838834764831845f * 1.4426950408889634f; // 1/sqrt(D)*log2e

    FragAB fq[4];
    #pragma unroll
    for (int f = 0; f < 4; ++f)
        #pragma unroll
        for (int j = 0; j < 8; ++j) {
            int feat = 32 * f + khalf * 16 + 2 * j;
            fq[f].u[j] = *(const unsigned*)(qB + (size_t)(bh * SLEN + qrow) * DHEAD + feat);
        }

    v8f accy[8] = {};
    float mrow[8], lrow[8];
    #pragma unroll
    for (int v = 0; v < 8; ++v) { mrow[v] = -1e30f; lrow[v] = 0.0f; }

    auto load_kv = [&](int t, int buf) {
        const int kv0 = t * 32;
        #pragma unroll
        for (int i = 0; i < 2; ++i) {          // K tile 32x128 bf16
            int p = tid + i * 256;
            int tok = p >> 4, c = p & 15;
            int swz = ((tok & 7) << 1) | ((tok & 8) >> 3);
            cp16_lds((char*)lK[buf] + tok * 256 + ((c ^ swz) * 16),
                     kB + (size_t)(bh * SLEN + kv0 + tok) * DHEAD + c * 8);
        }
        #pragma unroll
        for (int i = 0; i < 2; ++i) {          // V tile (pair-packed rows)
            int p = tid + i * 256;
            int r = p >> 5, c = p & 31;
            int cs = c ^ ((r >> 3) << 3);
            cp16_lds((char*)lV[buf] + r * 512 + cs * 16,
                     vtP + (size_t)(bh * (SLEN / 2) + (kv0 >> 1) + r) * DHEAD + c * 4);
        }
    };

    const int nt = q0 / 32 + 4;
    load_kv(0, 0);
    for (int t = 0; t < nt; ++t) {
        const int cur = t & 1;
        const int kv0 = t * 32;
        if (t + 1 < nt) { load_kv(t + 1, cur ^ 1); WAIT_ASYNC(4); }
        else            { WAIT_ASYNC(0); }
        __syncthreads();

        // ---- scores = q * K^T ----
        v8f sc[2] = {};
        #pragma unroll
        for (int ct = 0; ct < 2; ++ct) {
            int tok = ct * 16 + lane16;
            int swz = (((tok & 7) << 1) | ((tok & 8) >> 3)) << 2;
            #pragma unroll
            for (int f = 0; f < 4; ++f) {
                FragAB fb;
                #pragma unroll
                for (int j = 0; j < 8; ++j)
                    fb.u[j] = lK[cur][tok * 64 + ((f * 16 + khalf * 8 + j) ^ swz)];
                sc[ct] = __builtin_amdgcn_wmma_f32_16x16x32_bf16(
                    false, fq[f].v, false, fb.v, (short)0, sc[ct], false, false);
            }
        }

        // ---- causal mask + online softmax ----
        float pr0[8], pr1[8], rscale[8];
        const int myrowbase = q0 + wave * 16 + khalf * 8;
        #pragma unroll
        for (int v = 0; v < 8; ++v) {
            int qi = myrowbase + v;
            float s0 = sc[0][v] * sscale; if (kv0 + lane16 > qi)      s0 = -1e30f;
            float s1 = sc[1][v] * sscale; if (kv0 + 16 + lane16 > qi) s1 = -1e30f;
            float mx = fmaxf(s0, s1);
            mx = fmaxf(mx, __shfl_xor(mx, 1, 16));
            mx = fmaxf(mx, __shfl_xor(mx, 2, 16));
            mx = fmaxf(mx, __shfl_xor(mx, 4, 16));
            mx = fmaxf(mx, __shfl_xor(mx, 8, 16));
            float mnew = fmaxf(mrow[v], mx);
            float p0 = exp2f(s0 - mnew);
            float p1 = exp2f(s1 - mnew);
            float sum = p0 + p1;
            sum += __shfl_xor(sum, 1, 16);
            sum += __shfl_xor(sum, 2, 16);
            sum += __shfl_xor(sum, 4, 16);
            sum += __shfl_xor(sum, 8, 16);
            float rs = exp2f(mrow[v] - mnew);
            lrow[v] = lrow[v] * rs + sum;
            mrow[v] = mnew;
            rscale[v] = rs;
            pr0[v] = p0; pr1[v] = p1;
        }
        #pragma unroll
        for (int c = 0; c < 8; ++c)
            #pragma unroll
            for (int v = 0; v < 8; ++v)
                accy[c][v] *= rscale[v];

        // ---- P via per-wave LDS into A-fragment layout ----
        unsigned short* pp = lP[wave];
        #pragma unroll
        for (int v = 0; v < 8; ++v) {
            int r = khalf * 8 + v;
            pp[r * 40 + lane16]      = (unsigned short)bf16_1(pr0[v]);
            pp[r * 40 + 16 + lane16] = (unsigned short)bf16_1(pr1[v]);
        }
        FragAB fp;
        {
            const uint4* pa = (const uint4*)(pp + lane16 * 40 + khalf * 16);
            fp.q[0] = pa[0];
            fp.q[1] = pa[1];
        }

        // ---- Y += P * V ----
        #pragma unroll
        for (int c = 0; c < 8; ++c) {
            FragAB fv;
            #pragma unroll
            for (int j = 0; j < 8; ++j)
                fv.u[j] = lV[cur][(khalf * 8 + j) * 128 + ((c * 16 + lane16) ^ (khalf << 5))];
            accy[c] = __builtin_amdgcn_wmma_f32_16x16x32_bf16(
                false, fp.v, false, fv.v, (short)0, accy[c], false, false);
        }
        __syncthreads();
    }

    // ---- normalize, write y (bf16, heads re-interleaved [B,S,E]) ----
    float inv[8];
    #pragma unroll
    for (int v = 0; v < 8; ++v) inv[v] = 1.0f / lrow[v];
    #pragma unroll
    for (int c = 0; c < 8; ++c) {
        int feat = c * 16 + lane16;
        #pragma unroll
        for (int v = 0; v < 8; ++v) {
            int srow = q0 + wave * 16 + khalf * 8 + v;
            yB[((size_t)(b * SLEN + srow)) * EMB + h * DHEAD + feat] =
                (unsigned short)bf16_1(accy[c][v] * inv[v]);
        }
    }
}

// ============================================================================
extern "C" void kernel_launch(void* const* d_in, const int* in_sizes, int n_in,
                              void* d_out, int out_size, void* d_ws, size_t ws_size,
                              hipStream_t stream)
{
    const float* x  = (const float*)d_in[0];   // [B,S,E]
    const float* Wa = (const float*)d_in[1];   // [E,3E]
    const float* ba = (const float*)d_in[2];   // [3E]
    const float* Wp = (const float*)d_in[3];   // [E,E]
    const float* bp = (const float*)d_in[4];   // [E]

    const size_t tokE = (size_t)BATCH * SLEN * EMB;     // 16.7M elements
    float* out  = (float*)d_out;             // [B,S,E]
    float* kout = out + tokE;                // [B,H,S,D]
    float* vout = kout + tokE;               // [B,H,S,D]

    char* ws = (char*)d_ws;
    unsigned short* x_bf = (unsigned short*)ws;                          // 32 MB
    unsigned*       Wa_p = (unsigned*)(ws + 32ull * 1024 * 1024);        // 24 MB
    unsigned*       Wp_p = (unsigned*)(ws + 56ull * 1024 * 1024);        //  8 MB
    unsigned short* q_bf = (unsigned short*)(ws + 64ull * 1024 * 1024);  // 32 MB
    unsigned short* k_bf = (unsigned short*)(ws + 96ull * 1024 * 1024);  // 32 MB
    unsigned*       vt_p = (unsigned*)(ws + 128ull * 1024 * 1024);       // 32 MB
    unsigned short* y_bf = x_bf;             // overlay: x_bf dead after GEMM1

    dim3 blk(256);

    pack_adj_bf16<<<dim3((unsigned)(tokE / 2 / 256)), blk, 0, stream>>>(x, (unsigned*)x_bf);
    pack_pairs_bf16<<<dim3(3 * EMB / 256, EMB / 2), blk, 0, stream>>>(Wa, Wa_p, 3 * EMB);
    pack_pairs_bf16<<<dim3(EMB / 256, EMB / 2), blk, 0, stream>>>(Wp, Wp_p, EMB);

    // 1) qkv projection: scatter k,v fp32 (outputs) + q,k bf16 + packed V^T
    gemm_bf16_wmma<<<dim3(3 * EMB / 128, BATCH * SLEN / 128), blk, 0, stream>>>(
        x_bf, Wa_p, ba, nullptr, kout, vout, q_bf, k_bf, vt_p, 3 * EMB, EMB, 0);

    // 2) causal flash attention
    flash_attn_wmma<<<dim3(SLEN / 128, NHEAD, BATCH), blk, 0, stream>>>(
        q_bf, k_bf, vt_p, y_bf);

    // 3) output projection
    gemm_bf16_wmma<<<dim3(EMB / 128, BATCH * SLEN / 128), blk, 0, stream>>>(
        y_bf, Wp_p, bp, out, nullptr, nullptr, nullptr, nullptr, nullptr, EMB, EMB, 1);
}
